// EncoderBlock_48421461295290
// MI455X (gfx1250) — matrix operand order repaired and checked by reading, not compiled, not run on hardware
//
#include <hip/hip_runtime.h>
#include <hip/hip_bf16.h>

// ---------------------------------------------------------------------------
// CDNA5 (gfx1250, wave32) transformer encoder block.
// GEMMs + attention on v_wmma_f32_16x16x32_bf16; GEMM tiles staged by the
// Tensor Data Mover (tensor_load_to_lds, TENSORcnt) with LDS double buffering.
// ---------------------------------------------------------------------------

typedef __bf16 bf16_t;
typedef __attribute__((ext_vector_type(16))) __bf16 v16bf;
typedef __attribute__((ext_vector_type(8)))  __bf16 v8bf;
typedef __attribute__((ext_vector_type(8)))  float  v8f;
typedef __attribute__((ext_vector_type(4)))  unsigned int v4u;
typedef __attribute__((ext_vector_type(8)))  int v8i_t;
typedef __attribute__((ext_vector_type(4)))  int v4i_t;

#define TOK     8192     // B*L
#define DMODEL  768
#define NHEAD   12
#define DHEAD   64
#define FFN_D   3072
#define SEQ     1024
#define NBATCH  8

// Build a 16x32-bf16 half-fragment (32B per lane) from two 16B chunks.
__device__ __forceinline__ v16bf frag2(const bf16_t* p0, const bf16_t* p1) {
  union { v16bf v; v8bf h[2]; } u;
  u.h[0] = *(const v8bf*)p0;
  u.h[1] = *(const v8bf*)p1;
  return u.v;
}

__device__ __forceinline__ v8f wmma_bf16(v16bf a, v16bf b, v8f c) {
  // D = A(16x32) * B(32x16) + C, f32 accumulate
  return __builtin_amdgcn_wmma_f32_16x16x32_bf16(false, a, false, b,
                                                 (short)0, c, false, false);
}

__device__ __forceinline__ float gelu_exact(float x) {
  return 0.5f * x * (1.0f + erff(x * 0.70710678118654752f));
}

// ---------------------------------------------------------------------------
// TDM: 2-D tile load (bf16) global -> LDS, with LDS row padding of 8 elems
// per 64-elem row (matches the 72-element LDS row stride used below).
// D# layout per CDNA5 ISA ch.8. One call == one DMA, tracked on TENSORcnt.
// This toolchain exposes the 6-arg builtin: (v4u, v8i, v4i, v4i, v8i, i32).
// ---------------------------------------------------------------------------
__device__ __forceinline__ void tdm_load_tile_bf16(const bf16_t* gsrc,
                                                   unsigned widthElems,   // tensor dim0 (row length)
                                                   unsigned rowsTotal,    // tensor dim1
                                                   unsigned tileCols,     // tile dim0 (multiple of 64)
                                                   unsigned tileRows,     // tile dim1
                                                   unsigned ldsAddr)      // byte offset in LDS
{
  const unsigned long long ga = (unsigned long long)(size_t)gsrc;

  v4u g0;
  g0[0] = 1u;                                            // count=1, user D#
  g0[1] = ldsAddr;                                       // LDS dest (bytes)
  g0[2] = (unsigned)(ga & 0xffffffffull);                // global_addr[31:0]
  g0[3] = (unsigned)((ga >> 32) & 0x1ffffffull)          // global_addr[56:32]
          | (2u << 30);                                  // type=2 (image)

  v8i_t g1;
  g1[0] = (int)((1u << 16)        // data_size = 2 bytes
              | (1u << 20)        // pad_enable
              | (4u << 22)        // pad_interval: every 32 DWORDs (64 bf16)
              | (3u << 25));      // pad_amount: 4 DWORDs (8 bf16)
  g1[1] = (int)((widthElems & 0xffffu) << 16);                       // dim0 lo
  g1[2] = (int)(((widthElems >> 16) & 0xffffu)
              | ((rowsTotal & 0xffffu) << 16));                      // dim0 hi | dim1 lo
  g1[3] = (int)(((rowsTotal >> 16) & 0xffffu)
              | ((tileCols & 0xffffu) << 16));                       // dim1 hi | tile_dim0
  g1[4] = (int)(tileRows & 0xffffu);                                 // tile_dim1 (tile_dim2=0)
  g1[5] = (int)widthElems;                                           // dim0_stride[31:0]
  g1[6] = 0;                                                         // dim0_stride hi, dim1_stride lo
  g1[7] = 0;

  v4i_t gz4 = {0, 0, 0, 0};
  v8i_t gz8 = {0, 0, 0, 0, 0, 0, 0, 0};
  __builtin_amdgcn_tensor_load_to_lds(g0, g1, gz4, gz4, gz8, 0);
}

// ---------------------------------------------------------------------------
// f32 -> bf16 conversion (vectorized x4)
// ---------------------------------------------------------------------------
__global__ __launch_bounds__(256)
void cvt_f32_to_bf16(const float* __restrict__ x, bf16_t* __restrict__ y, int n4) {
  const int i = blockIdx.x * 256 + threadIdx.x;
  if (i >= n4) return;
  const float4 v = ((const float4*)x)[i];
  y[4 * i + 0] = (bf16_t)v.x;
  y[4 * i + 1] = (bf16_t)v.y;
  y[4 * i + 2] = (bf16_t)v.z;
  y[4 * i + 3] = (bf16_t)v.w;
}

// ---------------------------------------------------------------------------
// WMMA GEMM:  C[M,N] = A[M,K] @ W[N,K]^T (+bias, + optional epilogue)
//   EPI 0: bf16 out = acc + bias
//   EPI 1: bf16 out = gelu(acc + bias)
//   EPI 2: f32  out = acc + bias + res[row,col]
// BM=128, BN=64, BK=64; 256 threads = 8 waves on a 4x2 wave grid, 32x32/wave.
// Tiles staged by TDM into double-buffered LDS; wave 0 drives the DMA queue.
// ---------------------------------------------------------------------------
#define BM 128
#define BN 64
#define BK 64
#define LDSP (BK + 8)   // must match TDM pad: 64 elems + 8 pad

template <int EPI>
__global__ __launch_bounds__(256)
void gemm_wmma_bf16(const bf16_t* __restrict__ A, const bf16_t* __restrict__ Bw,
                    const float* __restrict__ bias, const float* __restrict__ res,
                    void* __restrict__ out, int M, int N, int K)
{
  __shared__ bf16_t sA[2][BM][LDSP];
  __shared__ bf16_t sB[2][BN][LDSP];

  const int tid    = threadIdx.x;
  const int wave   = tid >> 5;
  const int lane   = tid & 31;
  const int laneLo = lane & 15;
  const int laneHi = lane >> 4;

  const int bm = blockIdx.x * BM;
  const int bn = blockIdx.y * BN;
  const int wm = (wave >> 1) * 32;   // 0,32,64,96
  const int wn = (wave & 1) * 32;    // 0,32

  auto issue = [&](int buf, int k0) {
    tdm_load_tile_bf16(A + (size_t)bm * K + k0, (unsigned)K, (unsigned)M,
                       BK, BM, (unsigned)(size_t)&sA[buf][0][0]);
    tdm_load_tile_bf16(Bw + (size_t)bn * K + k0, (unsigned)K, (unsigned)N,
                       BK, BN, (unsigned)(size_t)&sB[buf][0][0]);
  };

  v8f acc[2][2] = {};

  if (wave == 0) issue(0, 0);                 // prologue DMA
  const int nk = K / BK;
  for (int kt = 0; kt < nk; ++kt) {
    const int cur = kt & 1;
    if (wave == 0) __builtin_amdgcn_s_wait_tensorcnt(0);   // cur buffer landed
    __syncthreads();
    if (wave == 0 && kt + 1 < nk) issue(cur ^ 1, (kt + 1) * BK);  // overlap

    #pragma unroll
    for (int ks = 0; ks < BK; ks += 32) {
      const int ka = ks + laneHi * 8;               // A-frag chunk base
      v16bf a0 = frag2(&sA[cur][wm      + laneLo][ka], &sA[cur][wm      + laneLo][ka + 16]);
      v16bf a1 = frag2(&sA[cur][wm + 16 + laneLo][ka], &sA[cur][wm + 16 + laneLo][ka + 16]);
      const int kb = ks + laneHi * 8;               // B-frag chunk base
      v16bf b0 = frag2(&sB[cur][wn      + laneLo][kb], &sB[cur][wn      + laneLo][kb + 16]);
      v16bf b1 = frag2(&sB[cur][wn + 16 + laneLo][kb], &sB[cur][wn + 16 + laneLo][kb + 16]);
      acc[0][0] = wmma_bf16(a0, b0, acc[0][0]);
      acc[0][1] = wmma_bf16(a0, b1, acc[0][1]);
      acc[1][0] = wmma_bf16(a1, b0, acc[1][0]);
      acc[1][1] = wmma_bf16(a1, b1, acc[1][1]);
    }
  }

  // epilogue: C/D layout -> lane owns col (lane&15), rows (lane>>4)*8 + r
  #pragma unroll
  for (int i = 0; i < 2; ++i) {
    #pragma unroll
    for (int j = 0; j < 2; ++j) {
      const int colg = bn + wn + j * 16 + laneLo;
      const float bv = bias[colg];
      #pragma unroll
      for (int r = 0; r < 8; ++r) {
        const int rowg = bm + wm + i * 16 + laneHi * 8 + r;
        float v = acc[i][j][r] + bv;
        if constexpr (EPI == 1) v = gelu_exact(v);
        if constexpr (EPI == 2) {
          v += res[(size_t)rowg * N + colg];
          ((float*)out)[(size_t)rowg * N + colg] = v;
        } else {
          ((bf16_t*)out)[(size_t)rowg * N + colg] = (bf16_t)v;
        }
      }
    }
  }
}

// ---------------------------------------------------------------------------
// Flash attention (non-causal), DH=64.  One block = (b, h, 128-query tile);
// 8 waves x 16 queries.  Online softmax, O accumulated in f32.
// ---------------------------------------------------------------------------
__global__ __launch_bounds__(256)
void flash_attn_wmma(const bf16_t* __restrict__ qg, const bf16_t* __restrict__ kg,
                     const bf16_t* __restrict__ vg, bf16_t* __restrict__ og)
{
  __shared__ bf16_t sK [64][72];      // [key][dh]
  __shared__ bf16_t sVT[64][72];      // [dh][key]  (V transposed for B-frags)
  __shared__ bf16_t sP [8][16][72];   // per-wave P scratch (C-layout -> A-layout)

  const int tid    = threadIdx.x;
  const int wave   = tid >> 5;
  const int lane   = tid & 31;
  const int laneLo = lane & 15;
  const int laneHi = lane >> 4;

  const int qt = blockIdx.x & 7;
  const int h  = (blockIdx.x >> 3) % NHEAD;
  const int b  = blockIdx.x / (8 * NHEAD);

  // Q fragments for this wave's 16 query rows (kept in registers)
  const int qrow = qt * 128 + wave * 16 + laneLo;
  const bf16_t* qp = qg + (size_t)(b * SEQ + qrow) * DMODEL + h * DHEAD;
  v16bf qf[2];
  #pragma unroll
  for (int s = 0; s < 2; ++s) {
    const int k0 = s * 32 + laneHi * 8;
    qf[s] = frag2(qp + k0, qp + k0 + 16);
  }

  float mrow[8], lrow[8];
  #pragma unroll
  for (int r = 0; r < 8; ++r) { mrow[r] = -1e30f; lrow[r] = 0.f; }
  v8f oacc[4] = {};

  const int ldRow = tid >> 2;          // 0..63 (key index in tile)
  const int ldCol = (tid & 3) * 16;    // dh chunk

  for (int kb = 0; kb < SEQ / 64; ++kb) {
    // cooperative K/V tile staging (+ prefetch of next tile)
    const int key = kb * 64 + ldRow;
    const bf16_t* kp = kg + (size_t)(b * SEQ + key) * DMODEL + h * DHEAD + ldCol;
    __builtin_prefetch(kp + (size_t)64 * DMODEL, 0, 0);
    v8bf kr0 = *(const v8bf*)kp, kr1 = *(const v8bf*)(kp + 8);
    const bf16_t* vp = vg + (size_t)(b * SEQ + key) * DMODEL + h * DHEAD + ldCol;
    __builtin_prefetch(vp + (size_t)64 * DMODEL, 0, 0);
    union { v8bf v; bf16_t e[8]; } vr0, vr1;
    vr0.v = *(const v8bf*)vp; vr1.v = *(const v8bf*)(vp + 8);

    *(v8bf*)&sK[ldRow][ldCol]     = kr0;
    *(v8bf*)&sK[ldRow][ldCol + 8] = kr1;
    #pragma unroll
    for (int e = 0; e < 8; ++e) {
      sVT[ldCol + e][ldRow]     = vr0.e[e];
      sVT[ldCol + 8 + e][ldRow] = vr1.e[e];
    }
    __syncthreads();

    // S = Q K^T   (16 queries x 64 keys, 8 WMMAs)
    v8f sAcc[4];
    #pragma unroll
    for (int nt = 0; nt < 4; ++nt) {
      const int col = nt * 16 + laneLo;     // key column for this lane
      v8f z = {};
      v16bf kb0 = frag2(&sK[col][laneHi * 8],      &sK[col][laneHi * 8 + 16]);
      v16bf kb1 = frag2(&sK[col][32 + laneHi * 8], &sK[col][32 + laneHi * 8 + 16]);
      z = wmma_bf16(qf[0], kb0, z);
      z = wmma_bf16(qf[1], kb1, z);
      sAcc[nt] = z;
    }

    // online softmax (half-wave reductions match the C-layout row split)
    const float sc  = 0.125f;              // 1/sqrt(DH)
    const float L2E = 1.44269504088896f;
    #pragma unroll
    for (int r = 0; r < 8; ++r) {
      float rm = sAcc[0][r];
      rm = fmaxf(rm, sAcc[1][r]); rm = fmaxf(rm, sAcc[2][r]); rm = fmaxf(rm, sAcc[3][r]);
      rm *= sc;
      #pragma unroll
      for (int off = 1; off < 16; off <<= 1)
        rm = fmaxf(rm, __shfl_xor(rm, off, 32));
      const float mn    = fmaxf(mrow[r], rm);
      const float alpha = exp2f((mrow[r] - mn) * L2E);
      mrow[r] = mn;
      lrow[r] *= alpha;
      #pragma unroll
      for (int nt = 0; nt < 4; ++nt) oacc[nt][r] *= alpha;
      float rs = 0.f;
      const int prow = laneHi * 8 + r;
      #pragma unroll
      for (int nt = 0; nt < 4; ++nt) {
        const float e = exp2f((sAcc[nt][r] * sc - mn) * L2E);
        rs += e;
        sP[wave][prow][nt * 16 + laneLo] = (bf16_t)e;   // wave-private, DS in-order
      }
      #pragma unroll
      for (int off = 1; off < 16; off <<= 1)
        rs += __shfl_xor(rs, off, 32);
      lrow[r] += rs;
    }

    // O += P V   (8 WMMAs)
    #pragma unroll
    for (int s = 0; s < 2; ++s) {
      const int ka = s * 32 + laneHi * 8;
      v16bf aP = frag2(&sP[wave][laneLo][ka], &sP[wave][laneLo][ka + 16]);
      #pragma unroll
      for (int nt = 0; nt < 4; ++nt) {
        const int dcol = nt * 16 + laneLo;      // dh column
        const int kv   = s * 32 + laneHi * 8;   // key contraction chunk
        v16bf bV = frag2(&sVT[dcol][kv], &sVT[dcol][kv + 16]);
        oacc[nt] = wmma_bf16(aP, bV, oacc[nt]);
      }
    }
    __syncthreads();
  }

  // normalize and write o[b, q, h*64 + dh] (bf16, feeds O-projection GEMM)
  #pragma unroll
  for (int r = 0; r < 8; ++r) {
    const float inv = 1.f / lrow[r];
    const int row = qt * 128 + wave * 16 + laneHi * 8 + r;
    bf16_t* op = og + (size_t)(b * SEQ + row) * DMODEL + h * DHEAD;
    #pragma unroll
    for (int nt = 0; nt < 4; ++nt)
      op[nt * 16 + laneLo] = (bf16_t)(oacc[nt][r] * inv);
  }
}

// ---------------------------------------------------------------------------
// Row LayerNorm over D=768; one row per block (256 threads, 8 waves).
// ---------------------------------------------------------------------------
template <bool OUT_BF16>
__global__ __launch_bounds__(256)
void layernorm_row(const float* __restrict__ x, const float* __restrict__ g,
                   const float* __restrict__ bb, void* __restrict__ y)
{
  const int row = blockIdx.x;
  const float* xr = x + (size_t)row * DMODEL;
  float s = 0.f, s2 = 0.f;
  for (int c = threadIdx.x; c < DMODEL; c += 256) {
    const float v = xr[c]; s += v; s2 += v * v;
  }
  #pragma unroll
  for (int off = 1; off < 32; off <<= 1) {
    s  += __shfl_xor(s,  off, 32);
    s2 += __shfl_xor(s2, off, 32);
  }
  __shared__ float red[2][8];
  const int wave = threadIdx.x >> 5, lane = threadIdx.x & 31;
  if (lane == 0) { red[0][wave] = s; red[1][wave] = s2; }
  __syncthreads();
  s = 0.f; s2 = 0.f;
  #pragma unroll
  for (int w = 0; w < 8; ++w) { s += red[0][w]; s2 += red[1][w]; }
  const float mu   = s  * (1.f / DMODEL);
  const float var  = s2 * (1.f / DMODEL) - mu * mu;
  const float rstd = rsqrtf(var + 1e-6f);
  for (int c = threadIdx.x; c < DMODEL; c += 256) {
    const float v = (xr[c] - mu) * rstd * g[c] + bb[c];
    if constexpr (OUT_BF16) ((bf16_t*)y)[(size_t)row * DMODEL + c] = (bf16_t)v;
    else                    ((float*)y)[(size_t)row * DMODEL + c]  = v;
  }
}

// ---------------------------------------------------------------------------
// Host-side pipeline
// ---------------------------------------------------------------------------
extern "C" void kernel_launch(void* const* d_in, const int* in_sizes, int n_in,
                              void* d_out, int out_size, void* d_ws, size_t ws_size,
                              hipStream_t stream)
{
  (void)in_sizes; (void)n_in; (void)out_size; (void)ws_size;

  const float* x    = (const float*)d_in[0];
  const float* x1   = (const float*)d_in[1];
  const float* Wq   = (const float*)d_in[2];
  const float* bq   = (const float*)d_in[3];
  const float* Wk   = (const float*)d_in[4];
  const float* bk   = (const float*)d_in[5];
  const float* Wv   = (const float*)d_in[6];
  const float* bv   = (const float*)d_in[7];
  const float* Wo   = (const float*)d_in[8];
  const float* bo   = (const float*)d_in[9];
  const float* W1   = (const float*)d_in[10];
  const float* b1   = (const float*)d_in[11];
  const float* W2   = (const float*)d_in[12];
  const float* b2   = (const float*)d_in[13];
  const float* ln1g = (const float*)d_in[14];
  const float* ln1b = (const float*)d_in[15];
  const float* ln2g = (const float*)d_in[16];
  const float* ln2b = (const float*)d_in[17];

  char* wsBase = (char*)d_ws;
  size_t off = 0;
  auto carve = [&](size_t bytes) -> void* {
    void* p = wsBase + off;
    off = (off + bytes + 255) & ~(size_t)255;
    return p;
  };

  bf16_t* xb   = (bf16_t*)carve((size_t)TOK * DMODEL * 2);
  bf16_t* x1b  = (bf16_t*)carve((size_t)TOK * DMODEL * 2);
  bf16_t* wqb  = (bf16_t*)carve((size_t)DMODEL * DMODEL * 2);
  bf16_t* wkb  = (bf16_t*)carve((size_t)DMODEL * DMODEL * 2);
  bf16_t* wvb  = (bf16_t*)carve((size_t)DMODEL * DMODEL * 2);
  bf16_t* wob  = (bf16_t*)carve((size_t)DMODEL * DMODEL * 2);
  bf16_t* w1b  = (bf16_t*)carve((size_t)FFN_D * DMODEL * 2);
  bf16_t* w2b  = (bf16_t*)carve((size_t)DMODEL * FFN_D * 2);
  bf16_t* qws  = (bf16_t*)carve((size_t)TOK * DMODEL * 2);
  bf16_t* kws  = (bf16_t*)carve((size_t)TOK * DMODEL * 2);
  bf16_t* vws  = (bf16_t*)carve((size_t)TOK * DMODEL * 2);
  bf16_t* ows  = (bf16_t*)carve((size_t)TOK * DMODEL * 2);
  float*  hws  = (float*) carve((size_t)TOK * DMODEL * 4);
  bf16_t* h2b  = (bf16_t*)carve((size_t)TOK * DMODEL * 2);
  bf16_t* f1ws = (bf16_t*)carve((size_t)TOK * FFN_D * 2);
  float*  h3ws = (float*) carve((size_t)TOK * DMODEL * 4);

  auto cvt = [&](const float* src, bf16_t* dst, size_t n) {
    const int n4 = (int)(n / 4);
    cvt_f32_to_bf16<<<dim3((n4 + 255) / 256), dim3(256), 0, stream>>>(src, dst, n4);
  };
  cvt(x,  xb,  (size_t)TOK * DMODEL);
  cvt(x1, x1b, (size_t)TOK * DMODEL);
  cvt(Wq, wqb, (size_t)DMODEL * DMODEL);
  cvt(Wk, wkb, (size_t)DMODEL * DMODEL);
  cvt(Wv, wvb, (size_t)DMODEL * DMODEL);
  cvt(Wo, wob, (size_t)DMODEL * DMODEL);
  cvt(W1, w1b, (size_t)FFN_D * DMODEL);
  cvt(W2, w2b, (size_t)DMODEL * FFN_D);

  const dim3 blk(256);
  const dim3 gD(TOK / BM, DMODEL / BN);   // 64 x 12
  const dim3 gF(TOK / BM, FFN_D / BN);    // 64 x 48

  // QKV projections (queries from x, keys/values from x1)
  gemm_wmma_bf16<0><<<gD, blk, 0, stream>>>(xb,  wqb, bq, nullptr, qws, TOK, DMODEL, DMODEL);
  gemm_wmma_bf16<0><<<gD, blk, 0, stream>>>(x1b, wkb, bk, nullptr, kws, TOK, DMODEL, DMODEL);
  gemm_wmma_bf16<0><<<gD, blk, 0, stream>>>(x1b, wvb, bv, nullptr, vws, TOK, DMODEL, DMODEL);

  // fused attention
  flash_attn_wmma<<<dim3(NBATCH * NHEAD * (SEQ / 128)), blk, 0, stream>>>(qws, kws, vws, ows);

  // output projection + residual (h = attn_out + x), f32
  gemm_wmma_bf16<2><<<gD, blk, 0, stream>>>(ows, wob, bo, x, hws, TOK, DMODEL, DMODEL);

  // post-norm FFN
  layernorm_row<true ><<<dim3(TOK), blk, 0, stream>>>(hws, ln2g, ln2b, h2b);
  gemm_wmma_bf16<1><<<gF, blk, 0, stream>>>(h2b,  w1b, b1, nullptr, f1ws, TOK, FFN_D, DMODEL);
  gemm_wmma_bf16<2><<<gD, blk, 0, stream>>>(f1ws, w2b, b2, hws,     h3ws, TOK, DMODEL, FFN_D);

  // final LayerNorm -> f32 output
  layernorm_row<false><<<dim3(TOK), blk, 0, stream>>>(h3ws, ln1g, ln1b, d_out);
}
